// SimpleModel_46566035423671
// MI455X (gfx1250) — matrix operand order repaired and checked by reading, not compiled, not run on hardware
//
#include <hip/hip_runtime.h>
#include <hip/hip_bf16.h>
#include <math.h>

// ---------------------------------------------------------------------------
// MACE SimpleModel forward for MI455X (gfx1250, wave32).
//   * Per-edge radial MLPs (8->64->64->{128,256}) run on v_wmma_f32_16x16x32_f16
//     (f16 inputs, f32 accumulation): 21.6 GFLOP total, weights pre-packed into
//     per-lane A-matrix tiles so each WMMA operand is one 32B load from L0/L2.
//   * Layer-to-layer C/D->B layout conversion done with shfl_xor(16) lane
//     exchanges + selects (wave-private, no LDS barriers, EXEC stays full).
//   * One wave <-> one 16-edge tile (straight-line, no grid-stride loop, so
//     the compiler doesn't keep all 44 A-tiles live across iterations).
//   * silu uses the native v_rcp_f32 (__builtin_amdgcn_rcpf) instead of IEEE
//     fp32 division: ~1 ulp, far below the f16 activation quantization.
//   * Message scatter uses global_atomic_add_f32 into L2-resident aggregates
//     (~1.3 GB L2 traffic -> the true roofline limiter; MLP hides under it).
//   * One-hot skip tensor-products collapse to 10-entry per-species tables.
// ---------------------------------------------------------------------------

typedef __attribute__((ext_vector_type(16))) _Float16 v16h;
typedef __attribute__((ext_vector_type(8)))  float    v8f;

__device__ __forceinline__ v8f wmma16(v16h a, v16h b, v8f c) {
  // D = A(16x32 f16) * B(32x16 f16) + C(16x16 f32)
  return __builtin_amdgcn_wmma_f32_16x16x32_f16(false, a, false, b, (short)0, c,
                                                false, false);
}

__device__ __forceinline__ float silu_f(float x) {
  // x * sigmoid(x) with native v_exp_f32 + v_rcp_f32 (no IEEE div chain)
  return x * __builtin_amdgcn_rcpf(1.0f + __expf(-x));
}

// Convert 4 C/D-layout f32 tiles (64 channels x 16 edges) into the two
// B-layout f16 operands (K=0..31 and K=32..63) for the next layer.
// C/D: lane l holds channel c = 16t + (l>>4)*8 + r, edge n = l&15.
// B:   lane l element j holds channel c = u*32 + (l>>4)*16 + j, edge n = l&15.
__device__ __forceinline__ void act_exchange(const v8f (&d)[4], int half,
                                             v16h& b0, v16h& b1) {
  float h[4][8], ex[4][8];
#pragma unroll
  for (int t = 0; t < 4; ++t)
#pragma unroll
    for (int r = 0; r < 8; ++r) h[t][r] = silu_f(d[t][r]);
#pragma unroll
  for (int t = 0; t < 4; ++t)
#pragma unroll
    for (int r = 0; r < 8; ++r) ex[t][r] = __shfl_xor(h[t][r], 16, 32);
#pragma unroll
  for (int u = 0; u < 2; ++u) {
    v16h b;
#pragma unroll
    for (int j = 0; j < 16; ++j) {
      float v = (j < 8) ? (half ? ex[2 * u + 1][j] : h[2 * u][j])
                        : (half ? h[2 * u + 1][j - 8] : ex[2 * u][j - 8]);
      b[j] = (_Float16)v;
    }
    if (u == 0) b0 = b; else b1 = b;
  }
}

// ---------------------------------------------------------------------------
// Prep kernels
// ---------------------------------------------------------------------------

// Per-species tables: su1_tab[z,d] = (W_embed[z] @ U1_s)[d]
//                     skip1_tab[z,d] = sum_u W_embed[z,u] * K1_s[u,z,d]
__global__ void prep_tabs(const float* __restrict__ We,
                          const float* __restrict__ U1s,
                          const float* __restrict__ K1s,
                          float* __restrict__ su1_tab,
                          float* __restrict__ skip1_tab) {
  int z = blockIdx.x, d = threadIdx.x;
  float su = 0.f, sk = 0.f;
  for (int c = 0; c < 64; ++c) {
    float w = We[z * 64 + c];
    su += w * U1s[c * 64 + d];
    sk += w * K1s[c * 640 + z * 64 + d];
  }
  su1_tab[z * 64 + d] = su;
  skip1_tab[z * 64 + d] = sk;
}

// Pack MLP weights (transposed) into per-lane WMMA A-matrix layout (f16).
// A element j of lane l: K = (l>>4)*16+j;
// logical k = kb*32+K; c = 16t + (l&15); value = W[k*Cout + c] (0 if k>=Kdim).
__global__ void prep_apack(const float* __restrict__ W1a,
                           const float* __restrict__ W2a,
                           const float* __restrict__ W3a,
                           const float* __restrict__ W1b,
                           const float* __restrict__ W2b,
                           const float* __restrict__ W3b,
                           unsigned short* __restrict__ apack) {
  int tid = blockIdx.x * blockDim.x + threadIdx.x;
  if (tid >= 72 * 512) return;
  int g = tid >> 9, lane = (tid >> 4) & 31, j = tid & 15;
  const float* W;
  int Kdim, Cout, t, kb;
  int gg = g;
  if (gg < 28) {  // interaction block 1: 4 + 8 + 16 tiles
    if (gg < 4)       { W = W1a; Kdim = 8;  Cout = 64;  t = gg;           kb = 0; }
    else if (gg < 12) { W = W2a; Kdim = 64; Cout = 64;  t = (gg - 4) >> 1;  kb = (gg - 4) & 1; }
    else              { W = W3a; Kdim = 64; Cout = 128; t = (gg - 12) >> 1; kb = (gg - 12) & 1; }
  } else {        // interaction block 2: 4 + 8 + 32 tiles
    gg -= 28;
    if (gg < 4)       { W = W1b; Kdim = 8;  Cout = 64;  t = gg;           kb = 0; }
    else if (gg < 12) { W = W2b; Kdim = 64; Cout = 64;  t = (gg - 4) >> 1;  kb = (gg - 4) & 1; }
    else              { W = W3b; Kdim = 64; Cout = 256; t = (gg - 12) >> 1; kb = (gg - 12) & 1; }
  }
  int half = lane >> 4, m = lane & 15;
  int K = half * 16 + j;
  int k = kb * 32 + K;
  float val = (k < Kdim) ? W[k * Cout + 16 * t + m] : 0.f;
  _Float16 hv = (_Float16)val;
  union { _Float16 f; unsigned short u; } cv;
  cv.f = hv;
  apack[tid] = cv.u;
}

__global__ void compute_species(const float* __restrict__ attrs,
                                int* __restrict__ species, int N) {
  int n = blockIdx.x * blockDim.x + threadIdx.x;
  if (n >= N) return;
  int best = 0;
  float bv = attrs[n * 10];
  for (int k = 1; k < 10; ++k) {
    float v = attrs[n * 10 + k];
    if (v > bv) { bv = v; best = k; }
  }
  species[n] = best;
}

// ---------------------------------------------------------------------------
// Edge geometry helper (per lane; both lanes of a pair redo the same edge)
// ---------------------------------------------------------------------------
__device__ __forceinline__ void edge_geom(const float* __restrict__ pos,
                                          const float* __restrict__ shifts,
                                          int snd, int rcv, int ec, bool valid,
                                          float& len, float& env,
                                          float& y1x, float& y1y, float& y1z) {
  float dx = pos[3 * snd + 0] - pos[3 * rcv + 0] + shifts[3 * ec + 0];
  float dy = pos[3 * snd + 1] - pos[3 * rcv + 1] + shifts[3 * ec + 1];
  float dz = pos[3 * snd + 2] - pos[3 * rcv + 2] + shifts[3 * ec + 2];
  len = sqrtf(dx * dx + dy * dy + dz * dz + 1e-12f);
  float inv = __builtin_amdgcn_rcpf(len);
  const float SQ3 = 1.7320508075688772f;
  y1x = SQ3 * dx * inv; y1y = SQ3 * dy * inv; y1z = SQ3 * dz * inv;
  float u = len * 0.2f;                       // len / r_max, r_max = 5
  float u2 = u * u, u4 = u2 * u2, u5 = u4 * u;
  float poly = 1.f - 21.f * u5 + 35.f * u5 * u - 15.f * u5 * u2;
  env = (valid && u < 1.f) ? 0.6324555320336759f * inv * poly : 0.f;  // sqrt(2/5)/len * cutoff
}

// ---------------------------------------------------------------------------
// Interaction block 1: fused radial-MLP (WMMA) + message construction +
// atomic scatter.  One wave <-> one tile of 16 edges; lanes l and l+16 share
// edge (l&15) and split its 64 channels by bit3.
// ---------------------------------------------------------------------------
__global__ __launch_bounds__(256)
void edge_block1(const float* __restrict__ pos, const int* __restrict__ ei,
                 const float* __restrict__ shifts,
                 const int* __restrict__ species,
                 const float* __restrict__ su1_tab,
                 const unsigned short* __restrict__ apackraw,
                 float* __restrict__ agg_s, float* __restrict__ agg_v,
                 int E, int numTiles) {
  const v16h* Ap = (const v16h*)apackraw;
  int lane = threadIdx.x & 31;
  int m = lane & 15, half = lane >> 4;
  int tile = blockIdx.x * (blockDim.x >> 5) + (threadIdx.x >> 5);
  if (tile >= numTiles) return;   // wave-uniform
  const v8f z8 = {0.f, 0.f, 0.f, 0.f, 0.f, 0.f, 0.f, 0.f};

  int e = tile * 16 + m;
  bool valid = (e < E);
  int ec = valid ? e : 0;
  int snd = ei[ec], rcv = ei[E + ec];
  float len, env, y1x, y1y, y1z;
  edge_geom(pos, shifts, snd, rcv, ec, valid, len, env, y1x, y1y, y1z);

  // B operand for layer 1: edge_feats^T, K padded 8->32 with zeros.
  v16h bx;
#pragma unroll
  for (int j = 0; j < 16; ++j) bx[j] = (_Float16)0.f;
  if (half == 0) {
#pragma unroll
    for (int j = 0; j < 8; ++j)
      bx[j] = (_Float16)(__sinf((float)(j + 1) * 0.6283185307179586f * len) * env);
  }

  // layer 1: 4 WMMAs
  v8f d[4];
#pragma unroll
  for (int t = 0; t < 4; ++t) d[t] = wmma16(Ap[t * 32 + lane], bx, z8);
  v16h b0, b1;
  act_exchange(d, half, b0, b1);

  // layer 2: 8 WMMAs
#pragma unroll
  for (int t = 0; t < 4; ++t) {
    v8f acc = wmma16(Ap[(4 + 2 * t) * 32 + lane], b0, z8);
    d[t] = wmma16(Ap[(4 + 2 * t + 1) * 32 + lane], b1, acc);
  }
  act_exchange(d, half, b0, b1);

  // layer 3 (16 WMMAs) fused with messages + atomic scatter
  int zs = valid ? species[snd] : 0;
#pragma unroll
  for (int t0 = 0; t0 < 4; ++t0) {
    v8f D0 = wmma16(Ap[(12 + 2 * t0) * 32 + lane], b0, z8);
    D0 = wmma16(Ap[(12 + 2 * t0 + 1) * 32 + lane], b1, D0);
    v8f D1 = wmma16(Ap[(12 + 2 * (t0 + 4)) * 32 + lane], b0, z8);
    D1 = wmma16(Ap[(12 + 2 * (t0 + 4) + 1) * 32 + lane], b1, D1);
    if (valid) {
#pragma unroll
      for (int r = 0; r < 8; ++r) {
        int c = 16 * t0 + half * 8 + r;
        float se = su1_tab[zs * 64 + c];     // s_e = (s0 @ U1_s) row of sender
        atomicAdd(&agg_s[rcv * 64 + c], D0[r] * se);
        float mv = D1[r] * se;
        atomicAdd(&agg_v[(rcv * 64 + c) * 3 + 0], mv * y1x);
        atomicAdd(&agg_v[(rcv * 64 + c) * 3 + 1], mv * y1y);
        atomicAdd(&agg_v[(rcv * 64 + c) * 3 + 2], mv * y1z);
      }
    }
  }
}

// ---------------------------------------------------------------------------
// Node update after block 1, fused with block-2 prep (one 64-thread block per
// node; s1/v1 never touch global memory).
// ---------------------------------------------------------------------------
__global__ __launch_bounds__(64)
void node_update(const float* __restrict__ agg_s1, const float* __restrict__ agg_v1,
                 const int* __restrict__ species,
                 const float* __restrict__ skip1_tab,
                 const float* __restrict__ L1s, const float* __restrict__ L1v,
                 const float* __restrict__ U2s, const float* __restrict__ U2v,
                 const float* __restrict__ K2s, const float* __restrict__ K2v,
                 float* __restrict__ s_up2, float* __restrict__ v_up2,
                 float* __restrict__ skip2_s, float* __restrict__ skip2_v) {
  __shared__ float sh_s[64];
  __shared__ float sh_v[64][3];
  int n = blockIdx.x, d = threadIdx.x;
  int z = species[n];
  float s1 = skip1_tab[z * 64 + d];
  float v1x = 0.f, v1y = 0.f, v1z = 0.f;
  for (int c = 0; c < 64; ++c) {
    float wls = L1s[c * 64 + d], wlv = L1v[c * 64 + d];
    s1 += agg_s1[n * 64 + c] * wls;
    v1x += agg_v1[(n * 64 + c) * 3 + 0] * wlv;
    v1y += agg_v1[(n * 64 + c) * 3 + 1] * wlv;
    v1z += agg_v1[(n * 64 + c) * 3 + 2] * wlv;
  }
  sh_s[d] = s1;
  sh_v[d][0] = v1x; sh_v[d][1] = v1y; sh_v[d][2] = v1z;
  __syncthreads();
  float su = 0.f, ks = 0.f;
  float vux = 0.f, vuy = 0.f, vuz = 0.f, kvx = 0.f, kvy = 0.f, kvz = 0.f;
  for (int c = 0; c < 64; ++c) {
    float sc = sh_s[c];
    float wu = U2s[c * 64 + d], wk = K2s[c * 640 + z * 64 + d];
    su += sc * wu; ks += sc * wk;
    float wv = U2v[c * 64 + d], wkv = K2v[c * 640 + z * 64 + d];
    float vx = sh_v[c][0], vy = sh_v[c][1], vz = sh_v[c][2];
    vux += vx * wv; vuy += vy * wv; vuz += vz * wv;
    kvx += vx * wkv; kvy += vy * wkv; kvz += vz * wkv;
  }
  int g = n * 64 + d;
  s_up2[g] = su;  skip2_s[g] = ks;
  v_up2[g * 3 + 0] = vux; v_up2[g * 3 + 1] = vuy; v_up2[g * 3 + 2] = vuz;
  skip2_v[g * 3 + 0] = kvx; skip2_v[g * 3 + 1] = kvy; skip2_v[g * 3 + 2] = kvz;
}

// ---------------------------------------------------------------------------
// Interaction block 2: fused radial-MLP (WMMA, 4 tpw groups) + 4 message
// paths (0e*0e, 0e*1o, 1o*0e, 1o*1o) + atomic scatter.
// ---------------------------------------------------------------------------
__global__ __launch_bounds__(256)
void edge_block2(const float* __restrict__ pos, const int* __restrict__ ei,
                 const float* __restrict__ shifts,
                 const float* __restrict__ s_up2, const float* __restrict__ v_up2,
                 const unsigned short* __restrict__ apackraw,
                 float* __restrict__ agg_s, float* __restrict__ agg_v,
                 int E, int numTiles) {
  const v16h* Ap = (const v16h*)apackraw;
  int lane = threadIdx.x & 31;
  int m = lane & 15, half = lane >> 4;
  int tile = blockIdx.x * (blockDim.x >> 5) + (threadIdx.x >> 5);
  if (tile >= numTiles) return;   // wave-uniform
  const v8f z8 = {0.f, 0.f, 0.f, 0.f, 0.f, 0.f, 0.f, 0.f};
  const float INV_SQ3 = 0.5773502691896258f;

  int e = tile * 16 + m;
  bool valid = (e < E);
  int ec = valid ? e : 0;
  int snd = ei[ec], rcv = ei[E + ec];
  float len, env, y1x, y1y, y1z;
  edge_geom(pos, shifts, snd, rcv, ec, valid, len, env, y1x, y1y, y1z);

  v16h bx;
#pragma unroll
  for (int j = 0; j < 16; ++j) bx[j] = (_Float16)0.f;
  if (half == 0) {
#pragma unroll
    for (int j = 0; j < 8; ++j)
      bx[j] = (_Float16)(__sinf((float)(j + 1) * 0.6283185307179586f * len) * env);
  }

  v8f d[4];
#pragma unroll
  for (int t = 0; t < 4; ++t) d[t] = wmma16(Ap[t * 32 + lane], bx, z8);
  v16h b0, b1;
  act_exchange(d, half, b0, b1);
#pragma unroll
  for (int t = 0; t < 4; ++t) {
    v8f acc = wmma16(Ap[(4 + 2 * t) * 32 + lane], b0, z8);
    d[t] = wmma16(Ap[(4 + 2 * t + 1) * 32 + lane], b1, acc);
  }
  act_exchange(d, half, b0, b1);

  // layer 3: 256 outputs = tpw[0..3][64]; tiles t0, t0+4, t0+8, t0+12
#pragma unroll
  for (int t0 = 0; t0 < 4; ++t0) {
    v8f DA = wmma16(Ap[(12 + 2 * t0) * 32 + lane], b0, z8);
    DA = wmma16(Ap[(12 + 2 * t0 + 1) * 32 + lane], b1, DA);
    v8f DB = wmma16(Ap[(12 + 2 * (t0 + 4)) * 32 + lane], b0, z8);
    DB = wmma16(Ap[(12 + 2 * (t0 + 4) + 1) * 32 + lane], b1, DB);
    v8f DC = wmma16(Ap[(12 + 2 * (t0 + 8)) * 32 + lane], b0, z8);
    DC = wmma16(Ap[(12 + 2 * (t0 + 8) + 1) * 32 + lane], b1, DC);
    v8f DD = wmma16(Ap[(12 + 2 * (t0 + 12)) * 32 + lane], b0, z8);
    DD = wmma16(Ap[(12 + 2 * (t0 + 12) + 1) * 32 + lane], b1, DD);
    if (valid) {
#pragma unroll
      for (int r = 0; r < 8; ++r) {
        int c = 16 * t0 + half * 8 + r;
        int gs = snd * 64 + c;
        float se = s_up2[gs];
        float vex = v_up2[gs * 3 + 0];
        float vey = v_up2[gs * 3 + 1];
        float vez = v_up2[gs * 3 + 2];
        float w0 = DA[r], w1 = DB[r], w2 = DC[r], w3 = DD[r];
        // mA: 0e x 0e -> 0e   (scalar channel c)
        atomicAdd(&agg_s[rcv * 128 + c], w0 * se);
        // mD: 1o x 1o -> 0e   (scalar channel 64+c)
        float dotv = vex * y1x + vey * y1y + vez * y1z;
        atomicAdd(&agg_s[rcv * 128 + 64 + c], w3 * dotv * INV_SQ3);
        // mB: 0e x 1o -> 1o   (vector channel c)
        float mb = w1 * se;
        atomicAdd(&agg_v[(rcv * 128 + c) * 3 + 0], mb * y1x);
        atomicAdd(&agg_v[(rcv * 128 + c) * 3 + 1], mb * y1y);
        atomicAdd(&agg_v[(rcv * 128 + c) * 3 + 2], mb * y1z);
        // mC: 1o x 0e -> 1o   (vector channel 64+c)
        atomicAdd(&agg_v[(rcv * 128 + 64 + c) * 3 + 0], w2 * vex);
        atomicAdd(&agg_v[(rcv * 128 + 64 + c) * 3 + 1], w2 * vey);
        atomicAdd(&agg_v[(rcv * 128 + 64 + c) * 3 + 2], w2 * vez);
      }
    }
  }
}

// ---------------------------------------------------------------------------
// Final node update + output assembly: out[n] = [s2(64) | v2.reshape(192)]
// ---------------------------------------------------------------------------
__global__ void node_final(const float* __restrict__ agg_s2,
                           const float* __restrict__ agg_v2,
                           const float* __restrict__ skip2_s,
                           const float* __restrict__ skip2_v,
                           const float* __restrict__ L2s,
                           const float* __restrict__ L2v,
                           float* __restrict__ out, int N) {
  int t = blockIdx.x * blockDim.x + threadIdx.x;
  if (t >= N * 64) return;
  int n = t >> 6, d = t & 63;
  float s2 = skip2_s[t];
  float vx = skip2_v[t * 3 + 0], vy = skip2_v[t * 3 + 1], vz = skip2_v[t * 3 + 2];
  for (int c = 0; c < 128; ++c) {
    float wls = L2s[c * 64 + d], wlv = L2v[c * 64 + d];
    s2 += agg_s2[n * 128 + c] * wls;
    vx += agg_v2[(n * 128 + c) * 3 + 0] * wlv;
    vy += agg_v2[(n * 128 + c) * 3 + 1] * wlv;
    vz += agg_v2[(n * 128 + c) * 3 + 2] * wlv;
  }
  out[n * 256 + d] = s2;
  out[n * 256 + 64 + d * 3 + 0] = vx;
  out[n * 256 + 64 + d * 3 + 1] = vy;
  out[n * 256 + 64 + d * 3 + 2] = vz;
}

// ---------------------------------------------------------------------------
extern "C" void kernel_launch(void* const* d_in, const int* in_sizes, int n_in,
                              void* d_out, int out_size, void* d_ws, size_t ws_size,
                              hipStream_t stream) {
  (void)n_in; (void)out_size; (void)ws_size;
  const float* pos    = (const float*)d_in[0];
  const float* attrs  = (const float*)d_in[1];
  const int*   ei     = (const int*)  d_in[2];
  const float* shifts = (const float*)d_in[3];
  const float* We     = (const float*)d_in[4];
  const float* R1W1   = (const float*)d_in[5];
  const float* R1W2   = (const float*)d_in[6];
  const float* R1W3   = (const float*)d_in[7];
  const float* U1s    = (const float*)d_in[8];
  const float* L1s    = (const float*)d_in[9];
  const float* L1v    = (const float*)d_in[10];
  const float* K1s    = (const float*)d_in[11];
  const float* R2W1   = (const float*)d_in[12];
  const float* R2W2   = (const float*)d_in[13];
  const float* R2W3   = (const float*)d_in[14];
  const float* U2s    = (const float*)d_in[15];
  const float* U2v    = (const float*)d_in[16];
  const float* L2s    = (const float*)d_in[17];
  const float* L2v    = (const float*)d_in[18];
  const float* K2s    = (const float*)d_in[19];
  const float* K2v    = (const float*)d_in[20];

  int N = in_sizes[0] / 3;
  int E = in_sizes[2] / 2;

  // ---- workspace carve-out (all 256B aligned) ----
  char* w = (char*)d_ws;
  size_t off = 0;
  auto take = [&](size_t bytes) -> char* {
    char* p = w + off;
    off = (off + bytes + 255) & ~(size_t)255;
    return p;
  };
  int*   species   = (int*)  take((size_t)N * 4);
  float* su1_tab   = (float*)take(640 * 4);
  float* skip1_tab = (float*)take(640 * 4);
  float* agg       = (float*)take((size_t)N * 768 * 4);   // all aggregates, one memset
  float* agg_s1 = agg;
  float* agg_v1 = agg + (size_t)N * 64;
  float* agg_s2 = agg + (size_t)N * 256;
  float* agg_v2 = agg + (size_t)N * 384;
  float* s_up2   = (float*)take((size_t)N * 64 * 4);
  float* v_up2   = (float*)take((size_t)N * 192 * 4);
  float* skip2_s = (float*)take((size_t)N * 64 * 4);
  float* skip2_v = (float*)take((size_t)N * 192 * 4);
  unsigned short* apack  = (unsigned short*)take(72 * 512 * 2);
  unsigned short* apack2 = apack + 28 * 512;

  hipMemsetAsync(agg, 0, (size_t)N * 768 * 4, stream);
  prep_tabs<<<10, 64, 0, stream>>>(We, U1s, K1s, su1_tab, skip1_tab);
  prep_apack<<<144, 256, 0, stream>>>(R1W1, R1W2, R1W3, R2W1, R2W2, R2W3, apack);
  compute_species<<<(N + 255) / 256, 256, 0, stream>>>(attrs, species, N);

  int tiles = (E + 15) / 16;
  int blocks = (tiles + 7) / 8;           // 8 waves (one tile each) per 256-thread block
  edge_block1<<<blocks, 256, 0, stream>>>(pos, ei, shifts, species, su1_tab,
                                          apack, agg_s1, agg_v1, E, tiles);
  node_update<<<N, 64, 0, stream>>>(agg_s1, agg_v1, species, skip1_tab,
                                    L1s, L1v, U2s, U2v, K2s, K2v,
                                    s_up2, v_up2, skip2_s, skip2_v);
  edge_block2<<<blocks, 256, 0, stream>>>(pos, ei, shifts, s_up2, v_up2,
                                          apack2, agg_s2, agg_v2, E, tiles);
  node_final<<<(N * 64 + 255) / 256, 256, 0, stream>>>(agg_s2, agg_v2, skip2_s,
                                                       skip2_v, L2s, L2v,
                                                       (float*)d_out, N);
}
